// SampGroup_26096221291156
// MI455X (gfx1250) — matrix operand order repaired and checked by reading, not compiled, not run on hardware
//
#include <hip/hip_runtime.h>

#define B_    16
#define N_    16384
#define SAM_  496
#define K_    64
#define OUTF_ 256
#define ROWS_ 1024          // B_*K_
#define EPS_  1e-5f

typedef _Float16 v16h __attribute__((ext_vector_type(16)));
typedef _Float16 v8h  __attribute__((ext_vector_type(8)));
typedef float    v8f  __attribute__((ext_vector_type(8)));

// ---------------------------------------------------------------------------
// Kernel 1: farthest point sampling. One block per batch. Points + running
// distances live in registers (16 per thread). idx[0]=0, then 495 update+argmax
// steps; argmax tie-break = lowest index (matches jnp.argmax first occurrence).
// ---------------------------------------------------------------------------
__global__ __launch_bounds__(1024) void fps_kernel(const float* __restrict__ x,
                                                   float* __restrict__ sp)
{
    const int b = blockIdx.x;
    const int t = threadIdx.x;
    const float* xb  = x  + (size_t)b * N_ * 3;
    float*       spb = sp + (size_t)b * SAM_ * 3;

    float px[16], py[16], pz[16], dist[16];
#pragma unroll
    for (int i = 0; i < 16; ++i) {
        int idx = t + i * 1024;
        px[i] = xb[idx * 3 + 0];
        py[i] = xb[idx * 3 + 1];
        pz[i] = xb[idx * 3 + 2];
        dist[i] = 1e10f;
    }

    __shared__ float rv[32];
    __shared__ int   ri[32];
    __shared__ int   curs;

    int cur = 0;
    if (t == 0) { spb[0] = xb[0]; spb[1] = xb[1]; spb[2] = xb[2]; }

    const int wv = t >> 5, ln = t & 31;

    for (int s = 1; s < SAM_; ++s) {
        float cx = xb[cur * 3 + 0];
        float cy = xb[cur * 3 + 1];
        float cz = xb[cur * 3 + 2];

        float bestv = -1.0f; int besti = N_;
#pragma unroll
        for (int i = 0; i < 16; ++i) {
            float dx = px[i] - cx, dy = py[i] - cy, dz = pz[i] - cz;
            float d  = dx * dx + dy * dy + dz * dz;
            float nd = fminf(dist[i], d);
            dist[i]  = nd;
            int idx  = t + i * 1024;
            if (nd > bestv || (nd == bestv && idx < besti)) { bestv = nd; besti = idx; }
        }
        // wave32 reduce (max val, min idx on tie)
#pragma unroll
        for (int off = 16; off > 0; off >>= 1) {
            float ov = __shfl_xor(bestv, off, 32);
            int   oi = __shfl_xor(besti, off, 32);
            if (ov > bestv || (ov == bestv && oi < besti)) { bestv = ov; besti = oi; }
        }
        if (ln == 0) { rv[wv] = bestv; ri[wv] = besti; }
        __syncthreads();
        if (wv == 0) {
            float v2 = rv[ln]; int i2 = ri[ln];
#pragma unroll
            for (int off = 16; off > 0; off >>= 1) {
                float ov = __shfl_xor(v2, off, 32);
                int   oi = __shfl_xor(i2, off, 32);
                if (ov > v2 || (ov == v2 && oi < i2)) { v2 = ov; i2 = oi; }
            }
            if (ln == 0) curs = i2;
        }
        __syncthreads();
        cur = curs;
        if (t == 0) {
            spb[s * 3 + 0] = xb[cur * 3 + 0];
            spb[s * 3 + 1] = xb[cur * 3 + 1];
            spb[s * 3 + 2] = xb[cur * 3 + 2];
        }
    }
}

// ---------------------------------------------------------------------------
// Kernel 2: convert W2 to f16 (WMMA B-operand source), row-major [n][c].
// ---------------------------------------------------------------------------
__global__ void cvt_kernel(const float* __restrict__ W2, _Float16* __restrict__ w2h)
{
    int i = blockIdx.x * blockDim.x + threadIdx.x;
    if (i < OUTF_ * OUTF_) w2h[i] = (_Float16)W2[i];
}

// ---------------------------------------------------------------------------
// Kernel 3: exact kNN (K=64 smallest, ascending, lower index on ties) among
// the 496 sampled points. One block of 128 threads per (b, s).
// ---------------------------------------------------------------------------
__global__ __launch_bounds__(128) void knn_kernel(const float* __restrict__ sp,
                                                  int* __restrict__ knn)
{
    const int s = blockIdx.x, b = blockIdx.y, t = threadIdx.x;
    __shared__ float d2[SAM_];
    __shared__ float rv[4];
    __shared__ int   ri[4];

    const float* spb = sp + (size_t)b * SAM_ * 3;
    float qx = spb[s * 3 + 0], qy = spb[s * 3 + 1], qz = spb[s * 3 + 2];
    for (int j = t; j < SAM_; j += 128) {
        float dx = spb[j * 3 + 0] - qx;
        float dy = spb[j * 3 + 1] - qy;
        float dz = spb[j * 3 + 2] - qz;
        d2[j] = dx * dx + dy * dy + dz * dz;
    }
    __syncthreads();

    int* row = knn + ((size_t)b * SAM_ + s) * K_;
    const int wv = t >> 5, ln = t & 31;
    for (int it = 0; it < K_; ++it) {
        float bv = 3.4e38f; int bi = SAM_;
        for (int j = t; j < SAM_; j += 128) {
            float v = d2[j];
            if (v < bv || (v == bv && j < bi)) { bv = v; bi = j; }
        }
#pragma unroll
        for (int off = 16; off > 0; off >>= 1) {
            float ov = __shfl_xor(bv, off, 32);
            int   oi = __shfl_xor(bi, off, 32);
            if (ov < bv || (ov == bv && oi < bi)) { bv = ov; bi = oi; }
        }
        if (ln == 0) { rv[wv] = bv; ri[wv] = bi; }
        __syncthreads();
        if (t == 0) {
            float v2 = rv[0]; int i2 = ri[0];
            for (int w = 1; w < 4; ++w)
                if (rv[w] < v2 || (rv[w] == v2 && ri[w] < i2)) { v2 = rv[w]; i2 = ri[w]; }
            row[it] = i2;
            d2[i2]  = 3.4e38f;
        }
        __syncthreads();
    }
}

// ---------------------------------------------------------------------------
// in-WG reduce of two f32 sums over 256 threads (8 waves)
// ---------------------------------------------------------------------------
__device__ __forceinline__ void reduce2(float& a, float& b, float* red, int t)
{
#pragma unroll
    for (int off = 16; off > 0; off >>= 1) {
        a += __shfl_xor(a, off, 32);
        b += __shfl_xor(b, off, 32);
    }
    int wv = t >> 5, ln = t & 31;
    if (ln == 0) { red[wv] = a; red[8 + wv] = b; }
    __syncthreads();
    if (t == 0) {
        float A = 0.f, Bv = 0.f;
        for (int i = 0; i < 8; ++i) { A += red[i]; Bv += red[8 + i]; }
        red[0] = A; red[8] = Bv;
    }
    __syncthreads();
    a = red[0]; b = red[8];
}

// ---------------------------------------------------------------------------
// Kernel 4: fused pipeline. One 256-thread WG per channel s (grid = 496).
// Pass A: GEMM1 (VALU f32, K=6) -> BN1 stats. Pass B: recompute h1, BN1+ReLU
// -> f16 LDS, WMMA GEMM2 -> BN2 stats. Pass C: same, normalize+ReLU, max over
// K via f32 D-layout + shfl + LDS int atomicMax (values >= 0 post-ReLU).
// ---------------------------------------------------------------------------
__global__ __launch_bounds__(256) void pipe_kernel(
    const float*    __restrict__ sp,     // [B,SAM,3]
    const int*      __restrict__ knn,    // [B,SAM,K]
    const _Float16* __restrict__ w2h,    // [256,256] f16 row-major [n][c]
    const float*    __restrict__ W1,     // [256,6]
    const float*    __restrict__ b1,     // [256]
    const float*    __restrict__ b2,     // [256]
    const float*    __restrict__ gamma_, // [496]
    const float*    __restrict__ beta_,  // [496]
    float*          __restrict__ out)    // [B,SAM,256]
{
    const int s  = blockIdx.x;
    const int t  = threadIdx.x;
    const int wv = t >> 5, ln = t & 31;

    __shared__ float feat[ROWS_][6];                         // 24.0 KB
    __shared__ __align__(16) _Float16 h1h[128][264];         // 66.0 KB (padded rows)
    __shared__ float w1s[OUTF_ * 6];                         //  6.0 KB
    __shared__ float b1s[OUTF_], b2s[OUTF_];                 //  2.0 KB
    __shared__ int   outacc[B_][OUTF_];                      // 16.0 KB
    __shared__ float redbuf[16];

    // ---- Phase 0: stage params, output accumulator, and feature rows ----
    for (int i = t; i < OUTF_ * 6; i += 256) w1s[i] = W1[i];
    b1s[t] = b1[t];
    b2s[t] = b2[t];
    for (int i = t; i < B_ * OUTF_; i += 256) ((int*)outacc)[i] = 0;

    for (int r = t; r < ROWS_; r += 256) {
        int b = r >> 6, k = r & 63;
        // faithful torch repeat/view shuffle: e=((b*SAM+s)*K+k); rem=e%(B*SAM)
        int e   = b * (SAM_ * K_) + s * K_ + k;
        int rem = e % (B_ * SAM_);
        int b2i = rem / SAM_;
        int s2  = rem % SAM_;
        const float* ps = sp + ((size_t)b2i * SAM_ + s2) * 3;
        float fx = ps[0], fy = ps[1], fz = ps[2];
        int g = knn[((size_t)b * SAM_ + s) * K_ + k];
        const float* pg = sp + ((size_t)b * SAM_ + g) * 3;
        feat[r][0] = fx;          feat[r][1] = fy;          feat[r][2] = fz;
        feat[r][3] = fx - pg[0];  feat[r][4] = fy - pg[1];  feat[r][5] = fz - pg[2];
    }
    const float gs = gamma_[s], bs = beta_[s];

    // ---- Per-wave B fragments of W2 (2 N-tiles x 8 K-steps), kept in regs.
    // B 32x16 f16 layout: lane = N (mod 16); lanes 0-15: K {k0..k0+7, k0+16..k0+23},
    // lanes 16-31: K {k0+8..k0+15, k0+24..k0+31}.
    v16h bfr[2][8];
    {
        const int laneN = ln & 15;
        const int ks    = (ln < 16) ? 0 : 8;
#pragma unroll
        for (int half = 0; half < 2; ++half) {
            int n = wv * 32 + half * 16 + laneN;
            const _Float16* wp = w2h + (size_t)n * 256;
#pragma unroll
            for (int kk = 0; kk < 8; ++kk) {
                const _Float16* p = wp + kk * 32 + ks;
                union { v16h v; v8h h[2]; } u;
                u.h[0] = *(const v8h*)(p);
                u.h[1] = *(const v8h*)(p + 16);
                bfr[half][kk] = u.v;
            }
        }
    }
    __syncthreads();

    // ---- Pass A: GEMM1 f32 + BN1 stats (thread t = output column t) ----
    float w1r[6];
#pragma unroll
    for (int j = 0; j < 6; ++j) w1r[j] = w1s[t * 6 + j];
    float bbA = b1s[t];

    float sum1 = 0.f, sq1 = 0.f;
    for (int r = 0; r < ROWS_; ++r) {
        float g = bbA;
#pragma unroll
        for (int j = 0; j < 6; ++j) g += feat[r][j] * w1r[j];
        sum1 += g;
        sq1  += g * g;
    }
    reduce2(sum1, sq1, redbuf, t);
    const float invN  = 1.0f / (float)(ROWS_ * OUTF_);
    float mean1 = sum1 * invN;
    float var1  = sq1 * invN - mean1 * mean1;
    float sc1   = gs * rsqrtf(var1 + EPS_);
    float sh1   = bs - mean1 * sc1;

    // regs for the 2-column h1 recompute (cols c0, c0+1; 64-row half rh)
    const int c0 = (t & 127) * 2;
    const int rh = t >> 7;
    float wa[6], wb[6];
#pragma unroll
    for (int j = 0; j < 6; ++j) { wa[j] = w1s[c0 * 6 + j]; wb[j] = w1s[(c0 + 1) * 6 + j]; }
    const float ba = b1s[c0], bbv = b1s[c0 + 1];

    auto computeH1 = [&](int blk) {
        int rbase = blk * 128 + rh * 64;
        for (int rr = 0; rr < 64; ++rr) {
            int r = rbase + rr;
            float f0 = feat[r][0], f1 = feat[r][1], f2 = feat[r][2];
            float f3 = feat[r][3], f4 = feat[r][4], f5 = feat[r][5];
            float ga = ba  + f0*wa[0] + f1*wa[1] + f2*wa[2] + f3*wa[3] + f4*wa[4] + f5*wa[5];
            float gb = bbv + f0*wb[0] + f1*wb[1] + f2*wb[2] + f3*wb[3] + f4*wb[4] + f5*wb[5];
            ga = fmaxf(ga * sc1 + sh1, 0.f);
            gb = fmaxf(gb * sc1 + sh1, 0.f);
            union { _Float16 h[2]; unsigned u; } pk;
            pk.h[0] = (_Float16)ga;
            pk.h[1] = (_Float16)gb;
            *(unsigned*)&h1h[rh * 64 + rr][c0] = pk.u;
        }
    };

    // A 16x32 f16 layout: lane 0-15: M=lane, K {kb..kb+7, kb+16..kb+23} with
    // kb=0; lanes 16-31: M=lane-16, kb=8.
    const int laneM = ln & 15;
    const int kb    = (ln < 16) ? 0 : 8;
    const int laneN = ln & 15;

    // ---- Pass B: WMMA GEMM2 -> BN2 stats ----
    float sum2 = 0.f, sq2 = 0.f;
    for (int blk = 0; blk < 8; ++blk) {
        __syncthreads();           // protect h1h from previous block's readers
        computeH1(blk);
        __syncthreads();
#pragma unroll 1
        for (int strip = 0; strip < 8; ++strip) {
            v8f acc0 = {}, acc1 = {};
#pragma unroll
            for (int kk = 0; kk < 8; ++kk) {
                const _Float16* p = &h1h[strip * 16 + laneM][kk * 32 + kb];
                union { v16h v; v8h h[2]; } u;
                u.h[0] = *(const v8h*)(p);
                u.h[1] = *(const v8h*)(p + 16);
                acc0 = __builtin_amdgcn_wmma_f32_16x16x32_f16(false, u.v, false, bfr[0][kk],
                                                              (short)0, acc0, false, false);
                acc1 = __builtin_amdgcn_wmma_f32_16x16x32_f16(false, u.v, false, bfr[1][kk],
                                                              (short)0, acc1, false, false);
            }
            float bb0 = b2s[wv * 32 + laneN];
            float bb1 = b2s[wv * 32 + 16 + laneN];
#pragma unroll
            for (int v = 0; v < 8; ++v) {
                float v0 = acc0[v] + bb0;
                float v1 = acc1[v] + bb1;
                sum2 += v0 + v1;
                sq2  += v0 * v0 + v1 * v1;
            }
        }
    }
    __syncthreads();
    reduce2(sum2, sq2, redbuf, t);
    float mean2 = sum2 * invN;
    float var2  = sq2 * invN - mean2 * mean2;
    float sc2   = gs * rsqrtf(var2 + EPS_);
    float sh2   = bs - mean2 * sc2;

    // ---- Pass C: recompute, normalize+ReLU, max over K ----
    for (int blk = 0; blk < 8; ++blk) {
        __syncthreads();
        computeH1(blk);
        __syncthreads();
#pragma unroll 1
        for (int strip = 0; strip < 8; ++strip) {
            v8f acc0 = {}, acc1 = {};
#pragma unroll
            for (int kk = 0; kk < 8; ++kk) {
                const _Float16* p = &h1h[strip * 16 + laneM][kk * 32 + kb];
                union { v16h v; v8h h[2]; } u;
                u.h[0] = *(const v8h*)(p);
                u.h[1] = *(const v8h*)(p + 16);
                acc0 = __builtin_amdgcn_wmma_f32_16x16x32_f16(false, u.v, false, bfr[0][kk],
                                                              (short)0, acc0, false, false);
                acc1 = __builtin_amdgcn_wmma_f32_16x16x32_f16(false, u.v, false, bfr[1][kk],
                                                              (short)0, acc1, false, false);
            }
            int   rtile = blk * 128 + strip * 16;
            int   bidx  = rtile >> 6;
            float bb0   = b2s[wv * 32 + laneN];
            float bb1   = b2s[wv * 32 + 16 + laneN];
            float m0 = 0.f, m1 = 0.f;   // post-ReLU values are >= 0
#pragma unroll
            for (int v = 0; v < 8; ++v) {
                float v0 = fmaxf((acc0[v] + bb0) * sc2 + sh2, 0.f);
                float v1 = fmaxf((acc1[v] + bb1) * sc2 + sh2, 0.f);
                m0 = fmaxf(m0, v0);
                m1 = fmaxf(m1, v1);
            }
            m0 = fmaxf(m0, __shfl_xor(m0, 16, 32));
            m1 = fmaxf(m1, __shfl_xor(m1, 16, 32));
            if (ln < 16) {
                atomicMax(&outacc[bidx][wv * 32 + laneN],      __float_as_int(m0));
                atomicMax(&outacc[bidx][wv * 32 + 16 + laneN], __float_as_int(m1));
            }
        }
    }
    __syncthreads();

    for (int i = t; i < B_ * OUTF_; i += 256) {
        int b = i >> 8, c = i & 255;
        out[((size_t)b * SAM_ + s) * OUTF_ + c] = __int_as_float(outacc[b][c]);
    }
}

// ---------------------------------------------------------------------------
// workspace layout: sp f32[16*496*3] @0 ; w2h f16[256*256] @95232 ;
// knn i32[16*496*64] @226304 ; total ~2.2 MB
// ---------------------------------------------------------------------------
extern "C" void kernel_launch(void* const* d_in, const int* in_sizes, int n_in,
                              void* d_out, int out_size, void* d_ws, size_t ws_size,
                              hipStream_t stream)
{
    const float* x   = (const float*)d_in[0];
    const float* W1  = (const float*)d_in[1];
    const float* b1v = (const float*)d_in[2];
    const float* W2  = (const float*)d_in[3];
    const float* b2v = (const float*)d_in[4];
    const float* gm  = (const float*)d_in[5];
    const float* bt  = (const float*)d_in[6];
    float* out = (float*)d_out;

    char* ws = (char*)d_ws;
    float*    sp  = (float*)(ws);
    _Float16* w2h = (_Float16*)(ws + 95232);
    int*      knn = (int*)(ws + 226304);

    fps_kernel<<<B_, 1024, 0, stream>>>(x, sp);
    cvt_kernel<<<(OUTF_ * OUTF_ + 255) / 256, 256, 0, stream>>>(W2, w2h);
    knn_kernel<<<dim3(SAM_, B_), 128, 0, stream>>>(sp, knn);
    pipe_kernel<<<SAM_, 256, 0, stream>>>(sp, knn, w2h, W1, b1v, b2v, gm, bt, out);
}
